// FixedConv1DPriorEnsemble_42580305773218
// MI455X (gfx1250) — matrix operand order repaired and checked by reading, not compiled, not run on hardware
//
#include <hip/hip_runtime.h>

// ---------------------------------------------------------------------------
// Fused 30-member conv1d ensemble for MI455X (gfx1250, wave32, WMMA).
//   x = embed[ids] * mask  -> 3 strided convs + relu -> mean pool -> head
//   out = sum_d z[d] * pi_d   [B=512, C=10]
//
// GEMM-ification (all K zero-padded in the A operand, so B pads need only be
// finite — B fragments are packed with UNconditional contiguous ds_load_b128):
//   conv1: shared input -> M=120(pad128), K=24(pad32)
//   conv2: members (2t,2t+1) block-diagonal in one 16-row A tile, K=24(pad32)
//   conv3: same pairing, K=48(pad64) -> two chained wmma K-steps
// Activations live in LDS as [position][channel] so B-fragment runs and
// C-tile stores are contiguous 16-byte LDS ops.  One WG per batch row.
// ---------------------------------------------------------------------------

typedef __attribute__((ext_vector_type(16))) _Float16 v16h;
typedef __attribute__((ext_vector_type(8)))  _Float16 v8h;
typedef __attribute__((ext_vector_type(8)))  float    v8f;

#define DZ      30
#define NPAIR   15
#define C_CLS   10
#define B_SZ    512
#define L_SEQ   4096
#define L3      1021
#define XPOS    296   // x positions per chunk: reads reach 2*143+3 = 289
#define H1POS   176   // h1 positions: reads reach 2*79+3 = 161 (pad zeroed once)
#define H1CH    128
#define H2POS   80    // h2 positions: reads reach 63+3 = 66
#define H2CH    256   // 240 used, padded for 512B row stride
#define NCHUNK  16    // ceil(1021 / 64)

struct __align__(16) Smem {
  _Float16 W1[128 * 32];         // [m=4d+oc][k=t*8+e]        (zero pad k>=24)
  _Float16 W2[NPAIR * 16 * 32];  // [pair][m][k=t*8+cin8]     (block-diag, pad)
  _Float16 W3[NPAIR * 16 * 64];  // [pair][m][k=t*16+cin16]   (block-diag, pad)
  float    b1v[128];
  float    b2v[240];
  float    b3v[240];
  _Float16 X [XPOS * 8];         // [pos][e]
  _Float16 H1[H1POS * H1CH];     // [pos][ch]
  _Float16 H2[H2POS * H2CH];     // [pos][ch]
  float    pool[240];
  float    head[DZ * C_CLS];
};

// A fragment 16x32 f16 (wave32): lanes 0-15 row M=lane hold K 0..15;
// lanes 16-31 row M=lane-16 hold K 16..31.  Two contiguous 8-half runs.
__device__ inline v16h load_a_frag(const _Float16* A, int lda, int lane) {
  const _Float16* p = A + (lane & 15) * lda + ((lane >> 4) * 16);
  union { v16h v; v8h h[2]; } u;
  u.h[0] = *(const v8h*)(p);
  u.h[1] = *(const v8h*)(p + 8);
  return u.v;
}
// B fragment 32x16 f16: lanes 0-15 col N=lane hold K=0..15 (element j = K),
// lanes 16-31 col N=lane-16 hold K=16..31.  Built from two contiguous runs.
__device__ inline v16h load_b_pair(const _Float16* p0, const _Float16* p1) {
  union { v16h v; v8h h[2]; } u;
  u.h[0] = *(const v8h*)p0;
  u.h[1] = *(const v8h*)p1;
  return u.v;
}

__global__ __launch_bounds__(256)
void fused_ensemble_kernel(const int* __restrict__ ids,
                           const float* __restrict__ mask,
                           const float* __restrict__ z,
                           const float* __restrict__ emb,
                           const float* __restrict__ W1g,
                           const float* __restrict__ b1g,
                           const float* __restrict__ W2g,
                           const float* __restrict__ b2g,
                           const float* __restrict__ W3g,
                           const float* __restrict__ b3g,
                           const float* __restrict__ Whg,
                           const float* __restrict__ bhg,
                           float* __restrict__ out) {
  extern __shared__ char smem_raw[];
  Smem& S = *reinterpret_cast<Smem*>(smem_raw);
  const int tid  = threadIdx.x;
  const int lane = tid & 31;
  const int wave = tid >> 5;          // 8 waves
  const int b    = blockIdx.x;
  const int col  = lane & 15;
  const int kb   = (lane >> 4) * 16;  // this lane's K base within a 32-K window

  // ---- preload weights into LDS as f16, zero-padded to WMMA shapes ----
  for (int i = tid; i < 128 * 32; i += 256) {
    const int m = i >> 5, k = i & 31;
    float v = 0.f;
    if (m < 120 && k < 24) {
      const int d = m >> 2, oc = m & 3, t = k >> 3, e = k & 7;
      v = W1g[((d * 4 + oc) * 8 + e) * 3 + t];
    }
    S.W1[i] = (_Float16)v;
  }
  for (int i = tid; i < NPAIR * 16 * 32; i += 256) {
    const int pp = i >> 9, r = i & 511, m = r >> 5, k = r & 31;
    const int t = k >> 3, cin = k & 7;
    float v = 0.f;
    if (t < 3) {
      if (m < 8)  { if (cin < 4)  v = W2g[(((2*pp)  *8 + m)     *4 + cin)    *3 + t]; }
      else        { if (cin >= 4) v = W2g[(((2*pp+1)*8 + (m-8)) *4 + (cin-4))*3 + t]; }
    }
    S.W2[i] = (_Float16)v;
  }
  for (int i = tid; i < NPAIR * 16 * 64; i += 256) {
    const int pp = i >> 10, r = i & 1023, m = r >> 6, k = r & 63;
    const int t = k >> 4, cin = k & 15;
    float v = 0.f;
    if (t < 3) {
      if (m < 8)  { if (cin < 8)  v = W3g[(((2*pp)  *8 + m)     *8 + cin)    *3 + t]; }
      else        { if (cin >= 8) v = W3g[(((2*pp+1)*8 + (m-8)) *8 + (cin-8))*3 + t]; }
    }
    S.W3[i] = (_Float16)v;
  }
  for (int i = tid; i < 128; i += 256) S.b1v[i] = (i < 120) ? b1g[i] : 0.f;
  for (int i = tid; i < 240; i += 256) S.b2v[i] = b2g[i];
  for (int i = tid; i < 240; i += 256) S.b3v[i] = b3g[i];
  // zero H1 pad region (positions 144..175) once: read by stage-2 halo taps
  for (int i = tid; i < (H1POS - 144) * (H1CH / 8); i += 256) {
    v8h zz = {};
    *(v8h*)&S.H1[144 * H1CH + i * 8] = zz;
  }

  // register pool accumulators: wave owns member-pairs pp = wave, wave+8
  float pacc[2][8];
#pragma unroll
  for (int q = 0; q < 2; ++q)
#pragma unroll
    for (int r = 0; r < 8; ++r) pacc[q][r] = 0.f;

  for (int c = 0; c < NCHUNK; ++c) {
    __syncthreads();  // protect X/H1/H2 reuse across chunks

    if (c + 1 < NCHUNK)  // pull next chunk's token ids toward L2/L0
      __builtin_prefetch(ids + (size_t)b * L_SEQ + 256 * (c + 1), 0, 1);

    // ---- gather + embed chunk: X[pos][e], pos 0..295, zero beyond L ----
    const int xg0 = 256 * c;
    for (int pos = tid; pos < XPOS; pos += 256) {
      const int xg = xg0 + pos;
      v8h row = {};
      if (xg < L_SEQ) {
        const int    id = ids [(size_t)b * L_SEQ + xg];
        const float  mk = mask[(size_t)b * L_SEQ + xg];
        const float4 e0 = ((const float4*)emb)[id * 2];
        const float4 e1 = ((const float4*)emb)[id * 2 + 1];
        row[0] = (_Float16)(e0.x * mk); row[1] = (_Float16)(e0.y * mk);
        row[2] = (_Float16)(e0.z * mk); row[3] = (_Float16)(e0.w * mk);
        row[4] = (_Float16)(e1.x * mk); row[5] = (_Float16)(e1.y * mk);
        row[6] = (_Float16)(e1.z * mk); row[7] = (_Float16)(e1.w * mk);
      }
      *(v8h*)&S.X[pos * 8] = row;
    }
    __syncthreads();

    // ---- stage 1: H1[pos][120] = relu(W1[120x24] @ im2col(X) + b1) ----
    {
      const int  mt = wave;  // one 16-row M tile per wave
      const v16h a  = load_a_frag(&S.W1[mt * 16 * 32], 32, lane);
      const int  rb = mt * 16 + ((lane >> 4) * 8);
      for (int nt = 0; nt < 9; ++nt) {
        const int colg = nt * 16 + col;
        // k = t*8 + e, t0 = kb/8 in {0,2}; t=3 run is A-zero-padded filler
        const _Float16* p0 = &S.X[(2 * colg + (kb >> 3)) * 8];
        const v16h bf = load_b_pair(p0, p0 + 8);
        v8f acc = {};
        acc = __builtin_amdgcn_wmma_f32_16x16x32_f16(false, a, false, bf,
                                                     (short)0, acc, false, false);
        v8h hrow;
#pragma unroll
        for (int r = 0; r < 8; ++r) {
          const float v = acc[r] + S.b1v[rb + r];
          hrow[r] = (_Float16)(v > 0.f ? v : 0.f);
        }
        *(v8h*)&S.H1[colg * H1CH + rb] = hrow;
      }
    }
    __syncthreads();

    // ---- stage 2: H2[pos][240] = relu(block-diag W2 @ im2col(H1) + b2) ----
    {
      const int mb = (lane >> 4) * 8;
#pragma unroll
      for (int q = 0; q < 2; ++q) {
        const int pp = wave + 8 * q;        // fixed pair ownership
        if (pp >= NPAIR) break;
        const v16h a = load_a_frag(&S.W2[pp * 16 * 32], 32, lane);
        for (int nt = 0; nt < 5; ++nt) {
          const int colg = nt * 16 + col;
          // k = t*8 + cin8; t0 = kb/8; t=3 run is A-zero-padded filler
          const _Float16* p0 = &S.H1[(2 * colg + (kb >> 3)) * H1CH + 8 * pp];
          const v16h bf = load_b_pair(p0, p0 + H1CH);
          v8f acc = {};
          acc = __builtin_amdgcn_wmma_f32_16x16x32_f16(false, a, false, bf,
                                                       (short)0, acc, false, false);
          v8h hrow;
#pragma unroll
          for (int r = 0; r < 8; ++r) {
            const float v = acc[r] + S.b2v[16 * pp + mb + r];
            hrow[r] = (_Float16)(v > 0.f ? v : 0.f);
          }
          *(v8h*)&S.H2[colg * H2CH + 16 * pp + mb] = hrow;
        }
      }
    }
    __syncthreads();

    // ---- stage 3: relu(block-diag W3 @ im2col(H2) + b3), pool in regs ----
    {
      const int mb = (lane >> 4) * 8;
#pragma unroll
      for (int q = 0; q < 2; ++q) {
        const int pp = wave + 8 * q;        // fixed pair ownership
        if (pp >= NPAIR) break;
        const v16h a0 = load_a_frag(&S.W3[pp * 16 * 64 +  0], 64, lane);
        const v16h a1 = load_a_frag(&S.W3[pp * 16 * 64 + 32], 64, lane);
        for (int nt = 0; nt < 4; ++nt) {
          const int colg = nt * 16 + col;
          v8f acc = {};
#pragma unroll
          for (int s = 0; s < 2; ++s) {     // K = 48 -> two chained K=32 steps
            // kk = t*16 + cin16; this lane-half's 16 elems share t = 2s+kb/16
            const int t = 2 * s + (kb >> 4);   // t=3 run is A-zero-padded filler
            const _Float16* p = &S.H2[(colg + t) * H2CH + 16 * pp];
            const v16h bf = load_b_pair(p, p + 8);
            acc = __builtin_amdgcn_wmma_f32_16x16x32_f16(
                false, (s == 0) ? a0 : a1, false, bf, (short)0, acc, false, false);
          }
          const int n3 = 64 * c + colg;     // global conv3 output position
          if (n3 < L3) {
#pragma unroll
            for (int r = 0; r < 8; ++r) {
              const float v = acc[r] + S.b3v[16 * pp + mb + r];
              pacc[q][r] += (v > 0.f ? v : 0.f);
            }
          }
        }
      }
    }
  }  // chunks

  // ---- reduce pooled sums across the 16 columns held by each half-wave ----
#pragma unroll
  for (int q = 0; q < 2; ++q) {
    const int pp = wave + 8 * q;
    if (pp < NPAIR) {
      const int mb = (lane >> 4) * 8;
#pragma unroll
      for (int r = 0; r < 8; ++r) {
        float v = pacc[q][r];
        v += __shfl_xor(v, 1);
        v += __shfl_xor(v, 2);
        v += __shfl_xor(v, 4);
        v += __shfl_xor(v, 8);              // masks < 16: stays in half-wave
        if (col == 0) S.pool[16 * pp + mb + r] = v;
      }
    }
  }
  __syncthreads();

  // ---- head: pi[d][cls] = mean(h3_d) @ Wh_d^T + bh_d ; out = sum_d z_d*pi ----
  const float inv = 1.0f / (float)L3;
  for (int idx = tid; idx < DZ * C_CLS; idx += 256) {
    const int d = idx / C_CLS;
    float acc = bhg[idx];
#pragma unroll
    for (int ch = 0; ch < 8; ++ch)
      acc += Whg[idx * 8 + ch] * (S.pool[d * 8 + ch] * inv);
    S.head[idx] = acc * z[d];
  }
  __syncthreads();
  if (tid < C_CLS) {
    float s = 0.f;
    for (int d = 0; d < DZ; ++d) s += S.head[d * C_CLS + tid];
    out[(size_t)b * C_CLS + tid] = s;
  }
}

extern "C" void kernel_launch(void* const* d_in, const int* in_sizes, int n_in,
                              void* d_out, int out_size, void* d_ws, size_t ws_size,
                              hipStream_t stream) {
  (void)in_sizes; (void)n_in; (void)d_ws; (void)ws_size; (void)out_size;
  const int*   ids  = (const int*)  d_in[0];
  const float* mask = (const float*)d_in[1];
  const float* z    = (const float*)d_in[2];
  const float* emb  = (const float*)d_in[3];
  const float* W1   = (const float*)d_in[4];
  const float* b1   = (const float*)d_in[5];
  const float* W2   = (const float*)d_in[6];
  const float* b2   = (const float*)d_in[7];
  const float* W3   = (const float*)d_in[8];
  const float* b3   = (const float*)d_in[9];
  const float* Wh   = (const float*)d_in[10];
  const float* bh   = (const float*)d_in[11];
  float* out = (float*)d_out;

  static_assert(sizeof(Smem) < 160 * 1024, "2 workgroups per 320KB WGP LDS");
  hipFuncSetAttribute(reinterpret_cast<const void*>(fused_ensemble_kernel),
                      hipFuncAttributeMaxDynamicSharedMemorySize,
                      (int)sizeof(Smem));
  fused_ensemble_kernel<<<dim3(B_SZ), dim3(256), sizeof(Smem), stream>>>(
      ids, mask, z, emb, W1, b1, W2, b2, W3, b3, Wh, bh, out);
}
